// LSTMLayer_61220463837880
// MI455X (gfx1250) — compile-verified, matrix operand order checked by reading, not run on hardware
//
#include <hip/hip_runtime.h>
#include <hip/hip_bf16.h>

typedef unsigned short u16;
typedef __attribute__((ext_vector_type(8)))  u16    v8s;
typedef __attribute__((ext_vector_type(16))) u16    v16s;
typedef __attribute__((ext_vector_type(16))) __bf16 v16bf;
typedef __attribute__((ext_vector_type(2)))  __bf16 v2bf;
typedef __attribute__((ext_vector_type(8)))  float  v8f;

#define B_DIM   4096
#define K_DIM   2048   // IN + OUT
#define N_DIM   1024   // OUT
#define BM      128    // rows per workgroup tile
#define BN      32     // cols per workgroup tile (per gate)
#define BK      32     // K per iteration (one bf16 WMMA K-step)
#define LDK     40     // padded K stride in LDS (u16 units): 80B, bank-friendly
#define ITERS   (K_DIM / BK)
#define A_HALF  (BM * LDK)          // u16 elements per A buffer
#define B_HALF  (4 * BN * LDK)      // u16 elements per B buffer

__device__ __forceinline__ u16 f2bf(float f) {
    // round-to-nearest-even f32 -> bf16
    unsigned u = __float_as_uint(f);
    unsigned r = u + 0x7FFFu + ((u >> 16) & 1u);
    return (u16)(r >> 16);
}

// pack two f32 into a dword of two bf16 (uses v_cvt_pk_bf16_f32 when available)
__device__ __forceinline__ unsigned pk2bf(float a, float b) {
#if __has_builtin(__builtin_amdgcn_cvt_pk_bf16_f32)
    v2bf r = __builtin_amdgcn_cvt_pk_bf16_f32(a, b);
    return __builtin_bit_cast(unsigned, r);
#else
    return (unsigned)f2bf(a) | ((unsigned)f2bf(b) << 16);
#endif
}

__device__ __forceinline__ float sigm(float x) {
    return 1.0f / (1.0f + __expf(-x));
}
__device__ __forceinline__ float tanh_fast(float x) {
    float e = __expf(2.0f * x);
    return (e - 1.0f) / (e + 1.0f);
}

__device__ __forceinline__ v16bf ld_frag(const u16* p) {
    v8s lo = *(const v8s*)(p);        // K 0..7  (lanes 0-15) / 8..15 (lanes 16-31)
    v8s hi = *(const v8s*)(p + 16);   // K 16..23 (lanes 0-15) / 24..31 (lanes 16-31)
    v16s cat = __builtin_shufflevector(lo, hi, 0,1,2,3,4,5,6,7,8,9,10,11,12,13,14,15);
    return __builtin_bit_cast(v16bf, cat);
}

__global__ __launch_bounds__(256)
void lstm_fused_wmma(const float* __restrict__ x,  const float* __restrict__ h,
                     const float* __restrict__ c,
                     const float* __restrict__ w_f, const float* __restrict__ b_f,
                     const float* __restrict__ w_i, const float* __restrict__ b_i,
                     const float* __restrict__ w_c, const float* __restrict__ b_c,
                     const float* __restrict__ w_o, const float* __restrict__ b_o,
                     float* __restrict__ out) {
    // double-buffered staging: ~40 KB of the 320 KB WGP LDS
    __shared__ __align__(16) u16 Alds[2 * A_HALF];   // [buf][row][k] bf16 bits
    __shared__ __align__(16) u16 Blds[2 * B_HALF];   // [buf][gate][n][k] bf16 bits (transposed)

    const int tid   = threadIdx.x;
    const int lane  = tid & 31;
    const int wv    = tid >> 5;            // 0..7 -> 16-row slice of the 128-row tile
    const int m16   = lane & 15;
    const int khalf = (lane >> 4) << 3;    // 0 or 8: ISA K-half per lane group
    const int nbase = blockIdx.x * BN;
    const int mbase = blockIdx.y * BM;

    const float* const W[4] = { w_f, w_i, w_c, w_o };

    v8f acc[4][2] = {};                    // [gate][coltile] f32 16x16 accumulators

    // B-staging coords: thread -> (k row, 4 consecutive n)
    const int kB = tid >> 3;               // 0..31
    const int n4 = (tid & 7) << 2;         // 0,4,...,28

    auto a_src = [&](int it) -> const float* {
        return (it * BK < 1024) ? x : h;   // z = [x | h]; tiles never straddle
    };

    auto stage = [&](int pbuf, int it) {
        const int k0 = it * BK;
        const float* asrc = a_src(it);
        const int kk = k0 & 1023;
        u16* Ab = Alds + pbuf * A_HALF;
        u16* Bb = Blds + pbuf * B_HALF;
        #pragma unroll
        for (int r = 0; r < 4; ++r) {
            int cid = tid + r * 256;       // 1024 chunks of 4 f32 (128 rows x 8 chunks)
            int row = cid >> 3;
            int kq  = (cid & 7) << 2;
            const float4 v = *(const float4*)(asrc + (size_t)(mbase + row) * 1024 + kk + kq);
            unsigned* dst = (unsigned*)&Ab[row * LDK + kq];
            dst[0] = pk2bf(v.x, v.y);
            dst[1] = pk2bf(v.z, v.w);
        }
        #pragma unroll
        for (int g = 0; g < 4; ++g) {
            const float4 v = *(const float4*)(W[g] + (size_t)(k0 + kB) * N_DIM + nbase + n4);
            u16* bp = Bb + g * (BN * LDK);
            bp[(n4 + 0) * LDK + kB] = f2bf(v.x);   // transpose at store time
            bp[(n4 + 1) * LDK + kB] = f2bf(v.y);
            bp[(n4 + 2) * LDK + kB] = f2bf(v.z);
            bp[(n4 + 3) * LDK + kB] = f2bf(v.w);
        }
    };

    auto compute = [&](int pbuf) {
        const u16* Ab = Alds + pbuf * A_HALF;
        const u16* Bb = Blds + pbuf * B_HALF;
        const v16bf afrag = ld_frag(&Ab[(wv * 16 + m16) * LDK + khalf]);
        #pragma unroll
        for (int g = 0; g < 4; ++g) {
            #pragma unroll
            for (int ct = 0; ct < 2; ++ct) {
                const v16bf bfrag =
                    ld_frag(&Bb[g * (BN * LDK) + (ct * 16 + m16) * LDK + khalf]);
                acc[g][ct] = __builtin_amdgcn_wmma_f32_16x16x32_bf16(
                    false, afrag, false, bfrag, (short)0, acc[g][ct], false, false);
            }
        }
    };

    // software pipeline: stage(0); { barrier; stage(next); compute(cur); }
    stage(0, 0);
    int p = 0;
    for (int it = 0; it < ITERS; ++it) {
        __syncthreads();                       // staging of buf p complete; buf p^1 free
        if (it + 1 < ITERS) {
            stage(p ^ 1, it + 1);
            if (it + 2 < ITERS) {              // warm L2 two tiles ahead (global_prefetch_b8)
                const int k2 = (it + 2) * BK;
                __builtin_prefetch(a_src(it + 2) + (size_t)(mbase + (tid >> 3)) * 1024 +
                                   (k2 & 1023) + ((tid & 7) << 2), 0, 1);
                #pragma unroll
                for (int g = 0; g < 4; ++g)
                    __builtin_prefetch(W[g] + (size_t)(k2 + kB) * N_DIM + nbase + n4, 0, 1);
            }
        }
        compute(p);
        p ^= 1;
    }

    // ---- fused LSTM gate epilogue ----
    // C/D layout: VGPR v -> row M = v + (lane>=16 ? 8 : 0), col N = lane&15
    const int rowbase = mbase + wv * 16 + khalf;   // khalf doubles as the +8 row offset
    #pragma unroll
    for (int ct = 0; ct < 2; ++ct) {
        const int n  = nbase + ct * 16 + m16;
        const float bf = b_f[n], bi = b_i[n], bc = b_c[n], bo = b_o[n];
        #pragma unroll
        for (int v = 0; v < 8; ++v) {
            const int r  = rowbase + v;
            const float cv = c[(size_t)r * N_DIM + n];
            const float fg = sigm(acc[0][ct][v] + bf);
            const float ig = sigm(acc[1][ct][v] + bi);
            const float gg = tanh_fast(acc[2][ct][v] + bc);
            const float og = sigm(acc[3][ct][v] + bo);
            const float cn = cv * fg + ig * gg;
            out[(size_t)r * N_DIM + n] = tanh_fast(cn) * og;
        }
    }
}

extern "C" void kernel_launch(void* const* d_in, const int* in_sizes, int n_in,
                              void* d_out, int out_size, void* d_ws, size_t ws_size,
                              hipStream_t stream) {
    const float* x   = (const float*)d_in[0];
    const float* h   = (const float*)d_in[1];
    const float* c   = (const float*)d_in[2];
    const float* w_f = (const float*)d_in[3];
    const float* b_f = (const float*)d_in[4];
    const float* w_i = (const float*)d_in[5];
    const float* b_i = (const float*)d_in[6];
    const float* w_c = (const float*)d_in[7];
    const float* b_c = (const float*)d_in[8];
    const float* w_o = (const float*)d_in[9];
    const float* b_o = (const float*)d_in[10];
    float* out = (float*)d_out;

    dim3 grid(N_DIM / BN, B_DIM / BM);   // 32 x 32 workgroups of 256 threads (8 waves)
    lstm_fused_wmma<<<grid, 256, 0, stream>>>(x, h, c, w_f, b_f, w_i, b_i,
                                              w_c, b_c, w_o, b_o, out);
}
